// MoeLayer_16192026705927
// MI455X (gfx1250) — matrix operand order repaired and checked by reading, not compiled, not run on hardware
//
#include <hip/hip_runtime.h>
#include <hip/hip_bf16.h>

// ---------------------------------------------------------------------------
// MoE top-2 FFN for MI455X (gfx1250, wave32, WMMA bf16 + TDM tile loads)
// ---------------------------------------------------------------------------
#define T_ 2048   // tokens (B*S)
#define D_ 1024   // model dim
#define E_ 8      // experts
#define F_ 4096   // ffn dim

typedef __attribute__((ext_vector_type(16))) __bf16 v16bf;
typedef __attribute__((ext_vector_type(8)))  __bf16 v8bf;
typedef __attribute__((ext_vector_type(8)))  float  v8f;
typedef __attribute__((ext_vector_type(4)))  unsigned int u32x4;
typedef __attribute__((ext_vector_type(8)))  int i32x8;
typedef __attribute__((ext_vector_type(4)))  int i32x4;

union BF16x16 { v16bf v; v8bf h8[2]; };

#if __has_builtin(__builtin_amdgcn_tensor_load_to_lds) && \
    __has_builtin(__builtin_amdgcn_s_wait_tensorcnt)
#define USE_TDM 1
#else
#define USE_TDM 0
#endif

__device__ __forceinline__ unsigned short f2bf(float f) {
    unsigned int u = __float_as_uint(f);
    u += 0x7FFFu + ((u >> 16) & 1u);        // round-to-nearest-even
    return (unsigned short)(u >> 16);
}

__device__ __forceinline__ unsigned lds_offset(const void* p) {
    // generic pointers to LDS carry the wave-relative offset in the low 32 bits
    return (unsigned)(unsigned long long)(uintptr_t)p;
}

#if USE_TDM
// Issue one TDM 2D tile load: tile_y rows x tile_x bf16 elements, row stride
// `stride` elements, packed contiguously into LDS at lds_off.
// D# bit layout per CDNA5 ISA ch.8 (group0 128b, group1 256b).
// This toolchain's builtin takes 6 args: (g0 u32x4, g1 i32x8, g2 i32x4,
// g3 i32x4, g4 i32x8, cpol). 2D tensors: groups 2..4 zero.
__device__ __forceinline__ void tdm_load_tile_2d(unsigned lds_off, const void* gsrc,
                                                 unsigned tile_x, unsigned tile_y,
                                                 unsigned long long stride) {
    unsigned long long ga = (unsigned long long)(uintptr_t)gsrc;
    u32x4 g0;
    g0[0] = 1u;                                              // count=1, user D#
    g0[1] = lds_off;                                         // lds_addr
    g0[2] = (unsigned)(ga & 0xFFFFFFFFu);                    // global_addr[31:0]
    g0[3] = (unsigned)((ga >> 32) & 0x1FFFFFFu) | (2u << 30); // [56:32], type=2
    i32x8 g1;
    g1[0] = (int)(1u << 16);                                 // data_size=1 -> 2B
    g1[1] = (int)((tile_x & 0xFFFFu) << 16);                 // tensor_dim0[15:0]
    g1[2] = (int)(((tile_x >> 16) & 0xFFFFu) |
                  ((tile_y & 0xFFFFu) << 16));               // dim0 hi | dim1 lo
    g1[3] = (int)(((tile_y >> 16) & 0xFFFFu) |
                  ((tile_x & 0xFFFFu) << 16));               // dim1 hi | tile_dim0
    g1[4] = (int)(tile_y & 0xFFFFu);                         // tile_dim1, tile_dim2=0
    g1[5] = (int)(stride & 0xFFFFFFFFu);                     // dim0_stride[31:0]
    g1[6] = (int)((stride >> 32) & 0xFFFFu);                 // dim0_stride[47:32]
    g1[7] = 0;
    i32x4 z4 = {0, 0, 0, 0};
    i32x8 z8 = {0, 0, 0, 0, 0, 0, 0, 0};
    __builtin_amdgcn_tensor_load_to_lds(g0, g1, z4, z4, z8, 0);
}
#endif

// --------------------------- x -> bf16 copy --------------------------------
__global__ void cvt_bf16_kernel(const float* __restrict__ x,
                                unsigned short* __restrict__ xb, int n4) {
    int i = blockIdx.x * blockDim.x + threadIdx.x;
    if (i >= n4) return;
    float4 p = ((const float4*)x)[i];
    ushort4 o;
    o.x = f2bf(p.x); o.y = f2bf(p.y); o.z = f2bf(p.z); o.w = f2bf(p.w);
    ((ushort4*)xb)[i] = o;
}

// ---------------- per-expert transpose + f32->bf16 convert ------------------
// w[e][R][C] f32  ->  wt[e][C][R] bf16 via coalesced 32x32 LDS tile transpose
__global__ void tcvt_kernel(const float* __restrict__ w,
                            unsigned short* __restrict__ wt, int R, int C) {
    __shared__ __align__(16) unsigned short tile[32][36];
    const int e  = blockIdx.z;
    const int r0 = blockIdx.y * 32, c0 = blockIdx.x * 32;
    const int tid = threadIdx.x;
    const int r  = tid >> 3, c4 = (tid & 7) * 4;
    float4 p = *(const float4*)&w[((size_t)e * R + r0 + r) * C + c0 + c4];
    tile[c4 + 0][r] = f2bf(p.x);
    tile[c4 + 1][r] = f2bf(p.y);
    tile[c4 + 2][r] = f2bf(p.z);
    tile[c4 + 3][r] = f2bf(p.w);
    __syncthreads();
    uint2 v = *(const uint2*)&tile[r][c4];
    *(uint2*)&wt[((size_t)e * C + c0 + r) * R + r0 + c4] = v;
}

// ------------------------------- gating ------------------------------------
__global__ void gate_kernel(const float* __restrict__ x,
                            const float* __restrict__ gw,
                            int* __restrict__ counts,
                            int* __restrict__ slots,
                            float* __restrict__ gwt) {
    const int lane = threadIdx.x & 31;
    const int t = blockIdx.x * (blockDim.x >> 5) + (threadIdx.x >> 5);
    if (t >= T_) return;
    float acc[E_];
#pragma unroll
    for (int e = 0; e < E_; ++e) acc[e] = 0.f;
    for (int d = lane; d < D_; d += 32) {
        float xv = x[t * D_ + d];
#pragma unroll
        for (int e = 0; e < E_; ++e) acc[e] += xv * gw[e * D_ + d];
    }
#pragma unroll
    for (int e = 0; e < E_; ++e) {
        for (int off = 16; off; off >>= 1)
            acc[e] += __shfl_xor(acc[e], off, 32);
    }
    if (lane == 0) {
        int e0 = 0;
#pragma unroll
        for (int e = 1; e < E_; ++e) if (acc[e] > acc[e0]) e0 = e;
        int e1 = (e0 == 0) ? 1 : 0;
#pragma unroll
        for (int e = 0; e < E_; ++e) if (e != e0 && acc[e] > acc[e1]) e1 = e;
        float m  = fmaxf(acc[e0], acc[e1]);
        float w0 = expf(acc[e0] - m);
        float w1 = expf(acc[e1] - m);
        float s  = w0 + w1; w0 /= s; w1 /= s;
        int p0 = atomicAdd(&counts[e0], 1);
        slots[e0 * T_ + p0] = t * 2 + 0;  gwt[e0 * T_ + p0] = w0;
        int p1 = atomicAdd(&counts[e1], 1);
        slots[e1 * T_ + p1] = t * 2 + 1;  gwt[e1 * T_ + p1] = w1;
    }
}

// --------------------- WMMA fragment helpers (LDS) -------------------------
// A fragment (16x32): lanes 0-15 row=lane, K 0..7 & 16..23; lanes 16-31
// row=lane-16, K 8..15 & 24..31 (documented interleaved layout).
__device__ __forceinline__ v16bf load_a_frag(const unsigned short* rowBase, int lane) {
    const int kb0 = (lane >> 4) * 8;
    BF16x16 u;
    u.h8[0] = *(const v8bf*)(rowBase + kb0);
    u.h8[1] = *(const v8bf*)(rowBase + kb0 + 16);
    return u.v;
}
// B fragment (32x16): lanes 0-15 col=lane hold K 0..7 & 16..23; lanes 16-31 K 8..15 & 24..31.
__device__ __forceinline__ v16bf load_b_frag(const unsigned short* colBase, int lane) {
    const int k0 = (lane >> 4) * 8;
    BF16x16 u;
    u.h8[0] = *(const v8bf*)(colBase + k0);
    u.h8[1] = *(const v8bf*)(colBase + k0 + 16);
    return u.v;
}

// ------------------------------ GEMM1 + SiLU --------------------------------
// h[slot, n] = silu( sum_d xb[token,d] * wt1[e,n,d] )   (wt1 pre-transposed)
// block = 256 thr = 8 waves -> 32(M) x 64(N) tile, K-tile 64 (2 WMMAs).
__global__ void gemm1_kernel(const unsigned short* __restrict__ xb,
                             const unsigned short* __restrict__ wt1,
                             const int* __restrict__ counts,
                             const int* __restrict__ slots,
                             unsigned short* __restrict__ h) {
    const int e   = blockIdx.z;
    const int cnt = counts[e];
    const int m0  = blockIdx.y * 32;
    if (m0 >= cnt) return;
    const int n0  = blockIdx.x * 64;

    __shared__ __align__(16) unsigned short lsA[32][64];   // [m][k]
    __shared__ __align__(16) unsigned short lsB[64][64];   // [n][k]
    __shared__ int rowSlot[32];

    const int tid = threadIdx.x;
    if (tid < 32) {
        int m = m0 + tid;
        rowSlot[tid] = (m < cnt) ? slots[e * T_ + m] : -1;
    }
    __syncthreads();

    const int wave = tid >> 5, lane = tid & 31;
    const int wm = wave >> 2, wn = wave & 3;
    const unsigned ldsB = lds_offset(&lsB[0][0]);
    v8f acc = {};

    for (int kb = 0; kb < D_; kb += 64) {
#if USE_TDM
        if (tid < 32)    // wave 0 issues one TDM 64x64 bf16 tile load for B
            tdm_load_tile_2d(ldsB, &wt1[((size_t)e * F_ + n0) * D_ + kb],
                             64u, 64u, (unsigned long long)D_);
#else
        {   // fallback: plain 16B copies (2 per thread)
            int r = tid >> 2, c = (tid & 3) * 16;
            const unsigned short* src = &wt1[((size_t)e * F_ + n0 + r) * D_ + kb + c];
            *(uint4*)&lsB[r][c]     = *(const uint4*)src;
            *(uint4*)&lsB[r][c + 8] = *(const uint4*)(src + 8);
        }
#endif
        // stage A: gathered token rows, one 16B copy per thread
        {
            int r = tid >> 3, c = (tid & 7) * 8;
            int s = rowSlot[r];
            uint4 av = make_uint4(0u, 0u, 0u, 0u);
            if (s >= 0)
                av = *(const uint4*)&xb[(size_t)(s >> 1) * D_ + kb + c];
            *(uint4*)&lsA[r][c] = av;
        }
#if USE_TDM
        if (tid < 32) __builtin_amdgcn_s_wait_tensorcnt(0);
#endif
        __syncthreads();
#pragma unroll
        for (int ko = 0; ko < 64; ko += 32) {
            v16bf a = load_a_frag(&lsA[wm * 16 + (lane & 15)][ko], lane);
            v16bf b = load_b_frag(&lsB[wn * 16 + (lane & 15)][ko], lane);
            acc = __builtin_amdgcn_wmma_f32_16x16x32_bf16(false, a, false, b,
                                                          (short)0, acc, false, false);
        }
        __syncthreads();
    }

    const int hsel = lane >> 4;
    const int col  = n0 + wn * 16 + (lane & 15);
#pragma unroll
    for (int r = 0; r < 8; ++r) {
        int mrow = wm * 16 + hsel * 8 + r;
        int s = rowSlot[mrow];
        if (s < 0) continue;
        float v = acc[r];
        float hv = v / (1.0f + expf(-v));            // SiLU
        h[(size_t)s * F_ + col] = f2bf(hv);
    }
}

// ------------------------------ GEMM2 + combine -----------------------------
// out[token, n] += gate_w * sum_f h[slot, f] * wt2[e, n, f]   (wt2 transposed)
__global__ void gemm2_kernel(const unsigned short* __restrict__ h,
                             const unsigned short* __restrict__ wt2,
                             const int* __restrict__ counts,
                             const int* __restrict__ slots,
                             const float* __restrict__ gwt,
                             float* __restrict__ out) {
    const int e   = blockIdx.z;
    const int cnt = counts[e];
    const int m0  = blockIdx.y * 32;
    if (m0 >= cnt) return;
    const int n0  = blockIdx.x * 64;

    __shared__ __align__(16) unsigned short lsA[32][64];
    __shared__ __align__(16) unsigned short lsB[64][64];
    __shared__ int   rowSlot[32];
    __shared__ float rowW[32];

    const int tid = threadIdx.x;
    if (tid < 32) {
        int m = m0 + tid;
        if (m < cnt) {
            rowSlot[tid] = slots[e * T_ + m];
            rowW[tid]    = gwt[e * T_ + m];
        } else {
            rowSlot[tid] = -1;
            rowW[tid]    = 0.f;
        }
    }
    __syncthreads();

    const int wave = tid >> 5, lane = tid & 31;
    const int wm = wave >> 2, wn = wave & 3;
    const unsigned ldsB = lds_offset(&lsB[0][0]);
    v8f acc = {};

    for (int kb = 0; kb < F_; kb += 64) {
#if USE_TDM
        if (tid < 32)
            tdm_load_tile_2d(ldsB, &wt2[((size_t)e * D_ + n0) * F_ + kb],
                             64u, 64u, (unsigned long long)F_);
#else
        {
            int r = tid >> 2, c = (tid & 3) * 16;
            const unsigned short* src = &wt2[((size_t)e * D_ + n0 + r) * F_ + kb + c];
            *(uint4*)&lsB[r][c]     = *(const uint4*)src;
            *(uint4*)&lsB[r][c + 8] = *(const uint4*)(src + 8);
        }
#endif
        {
            int r = tid >> 3, c = (tid & 7) * 8;
            int s = rowSlot[r];
            uint4 av = make_uint4(0u, 0u, 0u, 0u);
            if (s >= 0)
                av = *(const uint4*)&h[(size_t)s * F_ + kb + c];
            *(uint4*)&lsA[r][c] = av;
        }
#if USE_TDM
        if (tid < 32) __builtin_amdgcn_s_wait_tensorcnt(0);
#endif
        __syncthreads();
#pragma unroll
        for (int ko = 0; ko < 64; ko += 32) {
            v16bf a = load_a_frag(&lsA[wm * 16 + (lane & 15)][ko], lane);
            v16bf b = load_b_frag(&lsB[wn * 16 + (lane & 15)][ko], lane);
            acc = __builtin_amdgcn_wmma_f32_16x16x32_bf16(false, a, false, b,
                                                          (short)0, acc, false, false);
        }
        __syncthreads();
    }

    const int hsel = lane >> 4;
    const int col  = n0 + wn * 16 + (lane & 15);
#pragma unroll
    for (int r = 0; r < 8; ++r) {
        int mrow = wm * 16 + hsel * 8 + r;
        int s = rowSlot[mrow];
        if (s < 0) continue;
        int tok = s >> 1;
        atomicAdd(&out[(size_t)tok * D_ + col], rowW[mrow] * acc[r]);
    }
}

// ------------------------------- launcher ----------------------------------
extern "C" void kernel_launch(void* const* d_in, const int* in_sizes, int n_in,
                              void* d_out, int out_size, void* d_ws, size_t ws_size,
                              hipStream_t stream) {
    const float* x      = (const float*)d_in[0];   // [T, D]
    const float* gate_w = (const float*)d_in[1];   // [E, D]
    const float* w1     = (const float*)d_in[2];   // [E, D, F]
    const float* w2     = (const float*)d_in[3];   // [E, F, D]
    float* out          = (float*)d_out;           // [T, D]

    // workspace layout
    char* ws = (char*)d_ws;
    size_t off = 0;
    int*   counts = (int*)(ws + off);  off += 256;
    int*   slots  = (int*)(ws + off);  off += (size_t)E_ * T_ * 4;
    float* gwt    = (float*)(ws + off); off += (size_t)E_ * T_ * 4;
    unsigned short* xb   = (unsigned short*)(ws + off); off += (size_t)T_ * D_ * 2;
    unsigned short* hbuf = (unsigned short*)(ws + off); off += (size_t)T_ * 2 * F_ * 2;
    unsigned short* wt1  = (unsigned short*)(ws + off); off += (size_t)E_ * D_ * F_ * 2;
    unsigned short* wt2  = (unsigned short*)(ws + off); off += (size_t)E_ * D_ * F_ * 2;

    (void)hipMemsetAsync(counts, 0, E_ * sizeof(int), stream);
    (void)hipMemsetAsync(out, 0, (size_t)T_ * D_ * sizeof(float), stream);

    // activations -> bf16
    {
        int n4 = (T_ * D_) / 4;
        cvt_bf16_kernel<<<(n4 + 255) / 256, 256, 0, stream>>>(x, xb, n4);
    }
    // weights -> bf16, transposed so GEMM B-tiles are K-contiguous
    tcvt_kernel<<<dim3(F_ / 32, D_ / 32, E_), 256, 0, stream>>>(w1, wt1, D_, F_);
    tcvt_kernel<<<dim3(D_ / 32, F_ / 32, E_), 256, 0, stream>>>(w2, wt2, F_, D_);

    // gating + compaction (one wave per token)
    gate_kernel<<<T_ / 4, 128, 0, stream>>>(x, gate_w, counts, slots, gwt);

    // expert GEMM1 + SiLU
    gemm1_kernel<<<dim3(F_ / 64, T_ / 32, E_), 256, 0, stream>>>(
        xb, wt1, counts, slots, hbuf);

    // expert GEMM2 + weighted combine
    gemm2_kernel<<<dim3(D_ / 64, T_ / 32, E_), 256, 0, stream>>>(
        hbuf, wt2, counts, slots, gwt, out);

    (void)in_sizes; (void)n_in; (void)out_size; (void)ws_size;
}
